// TransformerEncoder_85512798864110
// MI455X (gfx1250) — matrix operand order repaired and checked by reading, not compiled, not run on hardware
//
#include <hip/hip_runtime.h>
#include <cstdint>

#define DMODEL 1024
#define FFDIM  2048
#define NB 4
#define NTT 16
#define NNN 197
#define NTOK (NB*NTT*NNN)   /* 12608 */
#define QKVW 1536
#define ATTN_SCALE 0.08838834764831845f

#ifndef __has_builtin
#define __has_builtin(x) 0
#endif
#if __has_builtin(__builtin_amdgcn_global_load_async_to_lds_b128)
#define HAVE_ASYNC_LDS 1
#else
#define HAVE_ASYNC_LDS 0
#endif

typedef __attribute__((ext_vector_type(16))) __bf16 v16bf;
typedef __attribute__((ext_vector_type(8)))  __bf16 v8bf;
typedef __attribute__((ext_vector_type(8)))  float  v8f;
typedef int vi4 __attribute__((vector_size(16)));

__device__ inline __bf16 f2bf(float f){
  unsigned u = __builtin_bit_cast(unsigned, f);
  unsigned r = u + 0x7FFFu + ((u >> 16) & 1u);
  unsigned short h = (unsigned short)(r >> 16);
  return __builtin_bit_cast(__bf16, h);
}
__device__ inline __bf16 bf_zero(){ unsigned short z = 0; return __builtin_bit_cast(__bf16, z); }

__device__ inline v16bf cat8(v8bf lo, v8bf hi){
  return __builtin_shufflevector(lo, hi, 0,1,2,3,4,5,6,7,8,9,10,11,12,13,14,15);
}
// A-fragment 16x32 bf16: rows = lane&15, K interleaved {0-7,16-23}|{8-15,24-31} per half-wave.
__device__ inline v16bf frag_a_rows(const __bf16* base, size_t rs, int row){
  int half = (threadIdx.x >> 4) & 1;
  const __bf16* p = base + (size_t)row * rs + half * 8;
  return cat8(*(const v8bf*)p, *(const v8bf*)(p + 16));
}
__device__ inline v16bf frag_a(const __bf16* base, size_t rs){
  return frag_a_rows(base, rs, threadIdx.x & 15);
}
// B-fragment 32x16 bf16 (source stored col-major: [n][k]): cols = lane&15, K interleaved {0-7,16-23}|{8-15,24-31} per half-wave.
__device__ inline v16bf frag_b_rows(const __bf16* base, size_t rs, int row){
  int half = (threadIdx.x >> 4) & 1;
  const __bf16* p = base + (size_t)row * rs + half * 8;
  return cat8(*(const v8bf*)p, *(const v8bf*)(p + 16));
}
__device__ inline v16bf frag_b(const __bf16* base, size_t rs){
  return frag_b_rows(base, rs, threadIdx.x & 15);
}
#define WMMA_BF16(A,B,C) __builtin_amdgcn_wmma_f32_16x16x32_bf16(false,(A),false,(B),(short)0,(C),false,false)

// ---------------- transpose + f32->bf16 weight conversion: wT[N][K] = w[K][N] ----------------
__global__ __launch_bounds__(256)
void k_cvt_t(const float* __restrict__ w, __bf16* __restrict__ wT, int K, int N){
  __shared__ __bf16 tile[32][33];
  int k0 = blockIdx.x * 32, n0 = blockIdx.y * 32;
  int tx = threadIdx.x & 31, ty = threadIdx.x >> 5;   // 32 x 8
#pragma unroll
  for (int j = 0; j < 4; ++j){
    int kk = ty + j * 8;
    tile[kk][tx] = f2bf(w[(size_t)(k0 + kk) * N + n0 + tx]);
  }
  __syncthreads();
#pragma unroll
  for (int j = 0; j < 4; ++j){
    int nn = ty + j * 8;
    wT[(size_t)(n0 + nn) * K + k0 + tx] = tile[tx][nn];
  }
}

// ---------------- LayerNorm -> bf16 ----------------
__global__ __launch_bounds__(256)
void k_ln(const float* __restrict__ x, const float* __restrict__ g,
          const float* __restrict__ bb, float eps, __bf16* __restrict__ out){
  __shared__ float red[16];
  int t = blockIdx.x, tid = threadIdx.x;
  const float* xr = x + (size_t)t * DMODEL;
  float v[4]; float s = 0.f, s2 = 0.f;
#pragma unroll
  for (int j = 0; j < 4; ++j){ v[j] = xr[tid + 256*j]; s += v[j]; s2 += v[j]*v[j]; }
#pragma unroll
  for (int d2 = 1; d2 < 32; d2 <<= 1){ s += __shfl_xor(s, d2, 32); s2 += __shfl_xor(s2, d2, 32); }
  int w = tid >> 5, lane = tid & 31;
  if (lane == 0){ red[w] = s; red[8 + w] = s2; }
  __syncthreads();
  if (tid == 0){ float a = 0.f, c = 0.f; for (int k = 0; k < 8; ++k){ a += red[k]; c += red[8+k]; } red[0] = a; red[8] = c; }
  __syncthreads();
  s = red[0]; s2 = red[8];
  float mean = s * (1.f / DMODEL);
  float inv  = rsqrtf(s2 * (1.f / DMODEL) - mean*mean + eps);
#pragma unroll
  for (int j = 0; j < 4; ++j){ int d2 = tid + 256*j;
    out[(size_t)t * DMODEL + d2] = f2bf((v[j] - mean) * inv * g[d2] + bb[d2]); }
}

// ---------------- x = LN(x + attn) in-place f32 ----------------
__global__ __launch_bounds__(256)
void k_add_ln(float* __restrict__ x, const float* __restrict__ attn,
              const float* __restrict__ g, const float* __restrict__ bb, float eps){
  __shared__ float red[16];
  int t = blockIdx.x, tid = threadIdx.x;
  float* xr = x + (size_t)t * DMODEL;
  const float* ar = attn + (size_t)t * DMODEL;
  float v[4]; float s = 0.f, s2 = 0.f;
#pragma unroll
  for (int j = 0; j < 4; ++j){ v[j] = xr[tid + 256*j] + ar[tid + 256*j]; s += v[j]; s2 += v[j]*v[j]; }
#pragma unroll
  for (int d2 = 1; d2 < 32; d2 <<= 1){ s += __shfl_xor(s, d2, 32); s2 += __shfl_xor(s2, d2, 32); }
  int w = tid >> 5, lane = tid & 31;
  if (lane == 0){ red[w] = s; red[8 + w] = s2; }
  __syncthreads();
  if (tid == 0){ float a = 0.f, c = 0.f; for (int k = 0; k < 8; ++k){ a += red[k]; c += red[8+k]; } red[0] = a; red[8] = c; }
  __syncthreads();
  s = red[0]; s2 = red[8];
  float mean = s * (1.f / DMODEL);
  float inv  = rsqrtf(s2 * (1.f / DMODEL) - mean*mean + eps);
#pragma unroll
  for (int j = 0; j < 4; ++j){ int d2 = tid + 256*j;
    xr[d2] = (v[j] - mean) * inv * g[d2] + bb[d2]; }
}

// ---------------- tiled WMMA GEMM: C[M][Nn] = A[M][K] @ Bt[Nn][K]^T ----------------
// 128x128 block tile, 8 waves (2M x 4N), wave tile 64x32, K step 32.
// A and B tiles share one LDS layout [row][k=32]; staged via async global->LDS when available.
__global__ __launch_bounds__(256)
void k_gemm(const __bf16* __restrict__ A, const __bf16* __restrict__ Bt,
            float* __restrict__ outF, __bf16* __restrict__ outH,
            const float* __restrict__ bias, const float* __restrict__ resid,
            int M, int Nn, int K, int relu){
  __shared__ __bf16 S[2][256 * 32];   // rows 0-127: A tile, rows 128-255: B tile; 16KB per buffer
  int tid = threadIdx.x, lane = tid & 31;
  int m0 = blockIdx.x * 128, n0 = blockIdx.y * 128;
  int wid = tid >> 5, wm = wid & 1, wn = wid >> 1;
  v8f acc[4][2];
#pragma unroll
  for (int i = 0; i < 4; ++i)
#pragma unroll
    for (int j = 0; j < 2; ++j) acc[i][j] = (v8f){0.f,0.f,0.f,0.f,0.f,0.f,0.f,0.f};

  // per-thread staging source row (thread tid stages LDS row tid: 64B = 4 x b128)
  int arow = m0 + tid; if (arow >= M) arow = M - 1;                 // clamp: pad rows never stored
  const __bf16* gsrc = (tid < 128) ? (A  + (size_t)arow * K)
                                   : (Bt + (size_t)(n0 + tid - 128) * K);
  const int nt = K / 32;

#if HAVE_ASYNC_LDS
  auto stage_async = [&](int buf, int kt){
    __bf16* lb = &S[buf][tid * 32];
#pragma unroll
    for (int j = 0; j < 4; ++j){
      __builtin_amdgcn_global_load_async_to_lds_b128(
          (vi4*)(gsrc + kt + j * 8), (vi4*)(lb + j * 8), 0, 0);
    }
  };
  stage_async(0, 0);
  for (int t = 0; t < nt; ++t){
    int buf = t & 1, kt = t * 32;
    if (t + 2 < nt) __builtin_prefetch(gsrc + kt + 64, 0, 1);
    if (t + 1 < nt){
      stage_async(buf ^ 1, kt + 32);
      asm volatile("s_wait_asynccnt 0x4" ::: "memory");   // drain stage t (in-order retire)
    } else {
      asm volatile("s_wait_asynccnt 0x0" ::: "memory");
    }
    __syncthreads();
    const __bf16* Sa = &S[buf][0];
    const __bf16* Sb = &S[buf][128 * 32];
    v16bf af[4], bfr[2];
#pragma unroll
    for (int mi = 0; mi < 4; ++mi) af[mi]  = frag_a(Sa + (wm*64 + mi*16) * 32, 32);
#pragma unroll
    for (int ni = 0; ni < 2; ++ni) bfr[ni] = frag_b(Sb + (wn*32 + ni*16) * 32, 32);
#pragma unroll
    for (int mi = 0; mi < 4; ++mi)
#pragma unroll
      for (int ni = 0; ni < 2; ++ni)
        acc[mi][ni] = WMMA_BF16(af[mi], bfr[ni], acc[mi][ni]);
    __syncthreads();                                      // free buf for stage t+2
  }
#else
  // fallback: register-prefetch software pipeline, single buffer
  v8bf r[4];
#pragma unroll
  for (int j = 0; j < 4; ++j) r[j] = *(const v8bf*)(gsrc + j * 8);
  for (int t = 0; t < nt; ++t){
    int kt = t * 32;
    __syncthreads();                                      // all waves done computing prev tile
#pragma unroll
    for (int j = 0; j < 4; ++j) *(v8bf*)(&S[0][tid * 32 + j * 8]) = r[j];
    if (t + 1 < nt){
#pragma unroll
      for (int j = 0; j < 4; ++j) r[j] = *(const v8bf*)(gsrc + kt + 32 + j * 8);
    }
    if (t + 2 < nt) __builtin_prefetch(gsrc + kt + 64, 0, 1);
    __syncthreads();
    const __bf16* Sa = &S[0][0];
    const __bf16* Sb = &S[0][128 * 32];
    v16bf af[4], bfr[2];
#pragma unroll
    for (int mi = 0; mi < 4; ++mi) af[mi]  = frag_a(Sa + (wm*64 + mi*16) * 32, 32);
#pragma unroll
    for (int ni = 0; ni < 2; ++ni) bfr[ni] = frag_b(Sb + (wn*32 + ni*16) * 32, 32);
#pragma unroll
    for (int mi = 0; mi < 4; ++mi)
#pragma unroll
      for (int ni = 0; ni < 2; ++ni)
        acc[mi][ni] = WMMA_BF16(af[mi], bfr[ni], acc[mi][ni]);
  }
#endif

  // epilogue: C layout -> VGPR i holds rows i (lanes 0-15) / i+8 (lanes 16-31), cols = lane&15
  int half = lane >> 4, c = lane & 15;
#pragma unroll
  for (int mi = 0; mi < 4; ++mi)
#pragma unroll
    for (int ni = 0; ni < 2; ++ni){
      int col = n0 + wn*32 + ni*16 + c;
      float bv = bias ? bias[col] : 0.f;
#pragma unroll
      for (int i = 0; i < 8; ++i){
        int row = m0 + wm*64 + mi*16 + i + 8*half;
        if (row < M){
          float v = acc[mi][ni][i] + bv;
          if (resid) v += resid[(size_t)row * Nn + col];
          if (relu)  v = fmaxf(v, 0.f);
          if (outH) outH[(size_t)row * Nn + col] = f2bf(v);
          else      outF[(size_t)row * Nn + col] = v;
        }
      }
    }
}

// ---------------- space attention: attend over N=197 within (b,t), 4 heads ----------------
__global__ __launch_bounds__(128)
void k_attn_space(const __bf16* __restrict__ qkv, float* __restrict__ attn){
  __shared__ __bf16 Psh[4][16 * 224];   // per-wave P (bf16 probs), 28KB
  __shared__ __bf16 VT[128 * 32];       // V transposed [dh][key] for current 32-key step, 8KB
  int tid = threadIdx.x, lane = tid & 31, w = tid >> 5;
  int g = blockIdx.x; int bt = g >> 2; int head = g & 3;
  int qt = blockIdx.y * 4 + w;
  bool valid = qt < 13;                 // wave-uniform
  int q0 = qt * 16;
  int c = lane & 15, half = lane >> 4;
  const __bf16* qb = qkv + (size_t)bt * NNN * QKVW + head * 128;
  const __bf16* kb = qb + 512;
  const __bf16* vb = qb + 1024;

  float rsum[8];
  if (valid){
    v16bf qf[4];
    int qrow = q0 + c; if (qrow > 196) qrow = 196;
#pragma unroll
    for (int s = 0; s < 4; ++s) qf[s] = frag_a_rows(qb + s*32, QKVW, qrow);
    v8f sc[14];
#pragma unroll 2
    for (int jb = 0; jb < 14; ++jb){
      v8f a = (v8f){0.f,0.f,0.f,0.f,0.f,0.f,0.f,0.f};
      int krow = jb*16 + c; bool oob = krow >= 197; if (krow > 196) krow = 196;
#pragma unroll
      for (int s = 0; s < 4; ++s){
        v16bf kf = frag_b_rows(kb + s*32, QKVW, krow);
        a = WMMA_BF16(qf[s], kf, a);
      }
#pragma unroll
      for (int i = 0; i < 8; ++i) a[i] = oob ? -1e30f : a[i] * ATTN_SCALE;
      sc[jb] = a;
    }
#pragma unroll
    for (int i = 0; i < 8; ++i){
      float m = -1e30f;
#pragma unroll
      for (int jb = 0; jb < 14; ++jb) m = fmaxf(m, sc[jb][i]);
#pragma unroll
      for (int d2 = 1; d2 < 16; d2 <<= 1) m = fmaxf(m, __shfl_xor(m, d2, 32));
      float ssum = 0.f;
#pragma unroll
      for (int jb = 0; jb < 14; ++jb){ float p = __expf(sc[jb][i] - m); sc[jb][i] = p; ssum += p; }
#pragma unroll
      for (int d2 = 1; d2 < 16; d2 <<= 1) ssum += __shfl_xor(ssum, d2, 32);
      rsum[i] = ssum;
    }
    __bf16* P = Psh[w];
#pragma unroll 2
    for (int jb = 0; jb < 14; ++jb)
#pragma unroll
      for (int i = 0; i < 8; ++i)
        P[(i + 8*half) * 224 + jb*16 + c] = f2bf(sc[jb][i]);
  }

  v8f accO[8];
#pragma unroll
  for (int i = 0; i < 8; ++i) accO[i] = (v8f){0.f,0.f,0.f,0.f,0.f,0.f,0.f,0.f};

  for (int ks = 0; ks < 7; ++ks){
    int kk = tid >> 2;                 // 0..31
    int dh0 = (tid & 3) * 32;
    int key = ks*32 + kk; if (key > 196) key = 196;
    const v8bf* sp = (const v8bf*)(vb + (size_t)key * QKVW + dh0);
    v8bf vv[4];
#pragma unroll
    for (int j = 0; j < 4; ++j) vv[j] = sp[j];
    __syncthreads();                   // prior-iter reads done (also covers P writes on iter 0)
#pragma unroll
    for (int j = 0; j < 4; ++j)
#pragma unroll
      for (int e = 0; e < 8; ++e)
        VT[(dh0 + j*8 + e) * 32 + kk] = vv[j][e];
    __syncthreads();
    if (valid){
      v16bf pf = frag_a(Psh[w] + ks*32, 224);
#pragma unroll
      for (int nt = 0; nt < 8; ++nt){
        v16bf vf = frag_b(VT + nt*16*32, 32);
        accO[nt] = WMMA_BF16(pf, vf, accO[nt]);
      }
    }
  }
  if (valid){
#pragma unroll
    for (int nt = 0; nt < 8; ++nt)
#pragma unroll
      for (int i = 0; i < 8; ++i){
        int row = q0 + i + 8*half;
        if (row < 197)
          attn[((size_t)bt * NNN + row) * DMODEL + head*128 + nt*16 + c] =
              accO[nt][i] * (1.f / rsum[i]);
      }
  }
}

// ---------------- time attention: attend over T=16 within (b,n), 4 heads ----------------
__global__ __launch_bounds__(128)
void k_attn_time(const __bf16* __restrict__ qkv, float* __restrict__ attn){
  __shared__ __bf16 Psh[4][16 * 32];    // 4KB
  __shared__ __bf16 VT[4][128 * 32];    // 32KB
  int tid = threadIdx.x, lane = tid & 31, h = tid >> 5;
  int bn = blockIdx.x; int b = bn / NNN; int n = bn % NNN;
  size_t rowbase = (size_t)b * NTT * NNN + n;
  size_t rs = (size_t)NNN * QKVW;
  const __bf16* qb = qkv + rowbase * QKVW + h * 128;
  int c = lane & 15, half = lane >> 4;

  v16bf qf[4];
#pragma unroll
  for (int s = 0; s < 4; ++s) qf[s] = frag_a(qb + s*32, rs);
  v8f sc = (v8f){0.f,0.f,0.f,0.f,0.f,0.f,0.f,0.f};
#pragma unroll
  for (int s = 0; s < 4; ++s){
    v16bf kf = frag_b(qb + 512 + s*32, rs);
    sc = WMMA_BF16(qf[s], kf, sc);
  }
  float rsum[8];
#pragma unroll
  for (int i = 0; i < 8; ++i){
    float vsc = sc[i] * ATTN_SCALE;
    float m = vsc;
#pragma unroll
    for (int d2 = 1; d2 < 16; d2 <<= 1) m = fmaxf(m, __shfl_xor(m, d2, 32));
    float p = __expf(vsc - m);
    float ssum = p;
#pragma unroll
    for (int d2 = 1; d2 < 16; d2 <<= 1) ssum += __shfl_xor(ssum, d2, 32);
    sc[i] = p; rsum[i] = ssum;
  }
  __bf16* P = Psh[h];
#pragma unroll
  for (int i = 0; i < 8; ++i) P[(i + 8*half) * 32 + c] = f2bf(sc[i]);
#pragma unroll
  for (int e = 0; e < 8; ++e) P[(lane & 15) * 32 + 16 + half*8 + e] = bf_zero();

  const __bf16* vbl = qkv + (rowbase + (size_t)c * NNN) * QKVW + 1024 + h*128 + half*64;
#pragma unroll
  for (int j = 0; j < 8; ++j){
    v8bf x = *(const v8bf*)(vbl + j*8);
#pragma unroll
    for (int e = 0; e < 8; ++e) VT[h][(half*64 + j*8 + e) * 32 + c] = x[e];
  }
#pragma unroll
  for (int d2 = 0; d2 < 4; ++d2)
#pragma unroll
    for (int e = 0; e < 16; ++e) VT[h][(lane*4 + d2) * 32 + 16 + e] = bf_zero();
  __syncthreads();

  v8f accO[8];
#pragma unroll
  for (int i = 0; i < 8; ++i) accO[i] = (v8f){0.f,0.f,0.f,0.f,0.f,0.f,0.f,0.f};
  v16bf pf = frag_a(Psh[h], 32);
#pragma unroll
  for (int nt = 0; nt < 8; ++nt){
    v16bf vf = frag_b(VT[h] + nt*16*32, 32);
    accO[nt] = WMMA_BF16(pf, vf, accO[nt]);
  }
#pragma unroll
  for (int nt = 0; nt < 8; ++nt)
#pragma unroll
    for (int i = 0; i < 8; ++i){
      size_t tok = ((size_t)(b*NTT + i + 8*half)) * NNN + n;
      attn[tok * DMODEL + 512 + h*128 + nt*16 + c] = accO[nt][i] * (1.f / rsum[i]);
    }
}

// ---------------- host orchestration ----------------
extern "C" void kernel_launch(void* const* d_in, const int* in_sizes, int n_in,
                              void* d_out, int out_size, void* d_ws, size_t ws_size,
                              hipStream_t stream){
  const float* src    = (const float*)d_in[0];
  const float* ln_s_g = (const float*)d_in[1];
  const float* ln_s_b = (const float*)d_in[2];
  const float* wqkv_s = (const float*)d_in[3];
  const float* ln_t_g = (const float*)d_in[4];
  const float* ln_t_b = (const float*)d_in[5];
  const float* wqkv_t = (const float*)d_in[6];
  const float* n1_g   = (const float*)d_in[7];
  const float* n1_b   = (const float*)d_in[8];
  const float* ff_g   = (const float*)d_in[9];
  const float* ff_b   = (const float*)d_in[10];
  const float* w1     = (const float*)d_in[11];
  const float* b1     = (const float*)d_in[12];
  const float* w2     = (const float*)d_in[13];
  const float* b2     = (const float*)d_in[14];
  float* x = (float*)d_out;

  auto al = [](size_t v){ return (v + 255) & ~(size_t)255; };
  size_t sWB  = al((size_t)DMODEL * FFDIM * 2);      // bf16 transposed-weight staging
  size_t sLN  = al((size_t)NTOK * DMODEL * 2);       // bf16 LN output
  size_t sQKV = al((size_t)NTOK * QKVW * 2);         // bf16 qkv
  size_t sAT  = al((size_t)NTOK * DMODEL * 4);       // f32 attn-out, reused as bf16 h[NTOK][2048]
  if (ws_size < sWB + sLN + sQKV + sAT) return;
  char* ws = (char*)d_ws;
  __bf16* wbuf = (__bf16*)ws;
  __bf16* lnb  = (__bf16*)(ws + sWB);
  __bf16* qkvb = (__bf16*)(ws + sWB + sLN);
  float*  attn = (float*) (ws + sWB + sLN + sQKV);
  __bf16* hbuf = (__bf16*)(ws + sWB + sLN + sQKV);

  (void)hipMemcpyAsync(x, src, (size_t)NTOK * DMODEL * sizeof(float),
                       hipMemcpyDeviceToDevice, stream);

  const int MB = (NTOK + 127) / 128;   // 99
  for (int l = 0; l < 4; ++l){
    // --- space branch ---
    k_cvt_t<<<dim3(DMODEL/32, QKVW/32), 256, 0, stream>>>(
        wqkv_s + (size_t)l*DMODEL*QKVW, wbuf, DMODEL, QKVW);
    k_ln<<<NTOK, 256, 0, stream>>>(x, ln_s_g + l*DMODEL, ln_s_b + l*DMODEL, 1e-5f, lnb);
    k_gemm<<<dim3(MB, QKVW/128), 256, 0, stream>>>(lnb, wbuf, nullptr, qkvb,
        nullptr, nullptr, NTOK, QKVW, DMODEL, 0);
    k_attn_space<<<dim3(NB*NTT*4, 4), 128, 0, stream>>>(qkvb, attn);
    // --- time branch ---
    k_cvt_t<<<dim3(DMODEL/32, QKVW/32), 256, 0, stream>>>(
        wqkv_t + (size_t)l*DMODEL*QKVW, wbuf, DMODEL, QKVW);
    k_ln<<<NTOK, 256, 0, stream>>>(x, ln_t_g + l*DMODEL, ln_t_b + l*DMODEL, 1e-5f, lnb);
    k_gemm<<<dim3(MB, QKVW/128), 256, 0, stream>>>(lnb, wbuf, nullptr, qkvb,
        nullptr, nullptr, NTOK, QKVW, DMODEL, 0);
    k_attn_time<<<NB*NNN, 128, 0, stream>>>(qkvb, attn);
    // --- residual + norm1 ---
    k_add_ln<<<NTOK, 256, 0, stream>>>(x, attn, n1_g + l*DMODEL, n1_b + l*DMODEL, 1e-5f);
    // --- FFN ---
    k_cvt_t<<<dim3(DMODEL/32, FFDIM/32), 256, 0, stream>>>(
        w1 + (size_t)l*DMODEL*FFDIM, wbuf, DMODEL, FFDIM);
    k_ln<<<NTOK, 256, 0, stream>>>(x, ff_g + l*DMODEL, ff_b + l*DMODEL, 1e-6f, lnb);
    k_gemm<<<dim3(MB, FFDIM/128), 256, 0, stream>>>(lnb, wbuf, nullptr, hbuf,
        b1 + (size_t)l*FFDIM, nullptr, NTOK, FFDIM, DMODEL, 1);
    k_cvt_t<<<dim3(FFDIM/32, DMODEL/32), 256, 0, stream>>>(
        w2 + (size_t)l*FFDIM*DMODEL, wbuf, FFDIM, DMODEL);
    k_gemm<<<dim3(MB, DMODEL/128), 256, 0, stream>>>(hbuf, wbuf, x, nullptr,
        b2 + (size_t)l*DMODEL, x, NTOK, DMODEL, FFDIM, 0);
  }
  (void)in_sizes; (void)n_in; (void)out_size;
}
